// MultiHeadedAttention_71425306132929
// MI455X (gfx1250) — matrix operand order repaired and checked by reading, not compiled, not run on hardware
//
#include <hip/hip_runtime.h>

// Problem constants (match reference)
constexpr int B_  = 2;
constexpr int S_  = 2048;
constexpr int DM_ = 1024;
constexpr int TD_ = 1024;
constexpr int H_  = 16;
constexpr int HD_ = 64;

typedef __attribute__((ext_vector_type(16))) _Float16 v16h;
typedef __attribute__((ext_vector_type(8)))  _Float16 v8h;
typedef __attribute__((ext_vector_type(8)))  float    v8f;

// ---------------------------------------------------------------------------
// WMMA fragment helpers (gfx1250 wave32; k order here is contiguous, A and B alike)
// A 16x32 f16 fragment, lane row m = lane&15:
//   lanes 0-15 : halves 0..15 = K0..15   (base = 0)
//   lanes 16-31: halves 0..15 = K16..31  (base = 8)
__device__ __forceinline__ v16h fragA(const _Float16* __restrict__ row, int base) {
    v8h lo = *(const v8h*)(row + 2 * base);
    v8h hi = *(const v8h*)(row + 2 * base + 8);
    v16h r;
#pragma unroll
    for (int i = 0; i < 8; ++i) { r[i] = lo[i]; r[8 + i] = hi[i]; }
    return r;
}
// B 32x16 f16 fragment for column n = lane&15: 16 contiguous K halfs starting
// at kbase = (lane<16 ? 0 : 16). Caller passes a pointer to those 16 halfs.
__device__ __forceinline__ v16h fragB(const _Float16* p) {
    v8h lo = *(const v8h*)(p);
    v8h hi = *(const v8h*)(p + 8);
    v16h r;
#pragma unroll
    for (int i = 0; i < 8; ++i) { r[i] = lo[i]; r[8 + i] = hi[i]; }
    return r;
}
__device__ __forceinline__ v8f wmma_f16(v16h a, v16h b, v8f c) {
    return __builtin_amdgcn_wmma_f32_16x16x32_f16(false, a, false, b, (short)0, c, false, false);
}

// ---------------------------------------------------------------------------
// CDNA5 async global->LDS copy (ASYNCcnt-tracked). 16B per lane per issue.
// vdst VGPR holds the LDS byte offset (low 32 bits of a generic shared ptr).
__device__ __forceinline__ void async_b128(const _Float16* gsrc, unsigned lds_off) {
    asm volatile("global_load_async_to_lds_b128 %0, %1, off"
                 :: "v"(lds_off), "v"(gsrc)
                 : "memory");
}
__device__ __forceinline__ void wait_async0() {
    asm volatile("s_wait_asynccnt 0x0" ::: "memory");
}
__device__ __forceinline__ unsigned lds_off(const void* p) {
    return (unsigned)(unsigned long long)p;   // low 32 bits = LDS offset
}

// ---------------------------------------------------------------------------
// shared 16x64-per-wave GEMM mainloop: acc(16x64) += A[16xK] * W[64xK]^T
__device__ __forceinline__ void gemm_mainloop(const _Float16* __restrict__ arow,
                                              const _Float16* __restrict__ w0,
                                              const _Float16* __restrict__ w1,
                                              const _Float16* __restrict__ w2,
                                              const _Float16* __restrict__ w3,
                                              int K, int abase,
                                              v8f& a0, v8f& a1, v8f& a2, v8f& a3) {
    for (int kb = 0; kb < K; kb += 32) {
        __builtin_prefetch(arow + kb + 128, 0, 0);
        v16h a  = fragA(arow + kb, abase);
        v16h b0 = fragB(w0 + kb);
        v16h b1 = fragB(w1 + kb);
        v16h b2 = fragB(w2 + kb);
        v16h b3 = fragB(w3 + kb);
        a0 = wmma_f16(a, b0, a0);
        a1 = wmma_f16(a, b1, a1);
        a2 = wmma_f16(a, b2, a2);
        a3 = wmma_f16(a, b3, a3);
    }
}

// ---------------------------------------------------------------------------
// 1) elementwise f32 -> f16
__global__ void cvt_f16_kernel(const float* __restrict__ in, _Float16* __restrict__ out, int n) {
    int i = blockIdx.x * blockDim.x + threadIdx.x;
    if (i < n) out[i] = (_Float16)in[i];
}

// ---------------------------------------------------------------------------
// 2) fused QKV GEMM + RoPE + head-split. One wave computes a 16x64 tile which
//    is exactly one head of one of {q,k,v} (64-col tiles are head-aligned).
//    RoPE pairs (d, d+32) live in (acc0,acc2)/(acc1,acc3) in the SAME lane, so
//    rotation is lane-local. Q gets the 1/sqrt(HD) scale folded in.
//    Outputs: Qr,Kr f16 [B,H,S,HD]; Vt f16 [B,H,HD,S] (transposed for P*V).
__global__ void __launch_bounds__(128)
gemm_qkv_rope_kernel(const _Float16* __restrict__ A, const _Float16* __restrict__ W,
                     const float* __restrict__ sinq, const float* __restrict__ cosq,
                     _Float16* __restrict__ Qr, _Float16* __restrict__ Kr,
                     _Float16* __restrict__ Vt) {
    const int lane  = threadIdx.x & 31;
    const int wave  = threadIdx.x >> 5;
    const int hi    = lane >> 4;
    const int ln    = lane & 15;
    const int abase = hi ? 8 : 0;
    const int kbase = hi ? 16 : 0;
    const int m0 = (blockIdx.y * 4 + wave) * 16;
    const int n0 = blockIdx.x * 64;

    const _Float16* arow = A + (size_t)(m0 + ln) * DM_;
    v8f acc0 = {}, acc1 = {}, acc2 = {}, acc3 = {};
    gemm_mainloop(arow,
                  W + (size_t)(n0      + ln) * DM_ + kbase,
                  W + (size_t)(n0 + 16 + ln) * DM_ + kbase,
                  W + (size_t)(n0 + 32 + ln) * DM_ + kbase,
                  W + (size_t)(n0 + 48 + ln) * DM_ + kbase,
                  DM_, abase, acc0, acc1, acc2, acc3);

    const int row0 = hi ? 8 : 0;
    const int seg  = n0 >> 10;            // 0=q, 1=k, 2=v
    const int hh   = (n0 & (TD_ - 1)) >> 6;
    if (seg == 2) {                       // V: write transposed [B,H,HD,S]
#pragma unroll
        for (int r = 0; r < 8; ++r) {
            const int srow = m0 + row0 + r;
            const int b = srow >> 11, s = srow & (S_ - 1);
            _Float16* base = Vt + ((size_t)(b * H_ + hh) * HD_) * S_ + s;
            base[(size_t)(ln)      * S_] = (_Float16)acc0[r];
            base[(size_t)(16 + ln) * S_] = (_Float16)acc1[r];
            base[(size_t)(32 + ln) * S_] = (_Float16)acc2[r];
            base[(size_t)(48 + ln) * S_] = (_Float16)acc3[r];
        }
    } else {                              // Q or K: lane-local RoPE
        _Float16* dstT = (seg == 0) ? Qr : Kr;
        const float sc = (seg == 0) ? 0.125f : 1.0f;   // 1/sqrt(64) into Q
#pragma unroll
        for (int r = 0; r < 8; ++r) {
            const int srow = m0 + row0 + r;
            const int b = srow >> 11, s = srow & (S_ - 1);
            const float* cs = cosq + s * HD_;
            const float* sn = sinq + s * HD_;
            const float c0 = cs[ln], c1 = cs[16 + ln], c2 = cs[32 + ln], c3 = cs[48 + ln];
            const float s0 = sn[ln], s1 = sn[16 + ln], s2 = sn[32 + ln], s3 = sn[48 + ln];
            const float x0 = acc0[r], x1 = acc1[r], x2 = acc2[r], x3 = acc3[r];
            const float y0 = (x0 * c0 - x2 * s0) * sc;   // d<32 : -x[d+32]*sin
            const float y1 = (x1 * c1 - x3 * s1) * sc;
            const float y2 = (x2 * c2 + x0 * s2) * sc;   // d>=32: +x[d-32]*sin
            const float y3 = (x3 * c3 + x1 * s3) * sc;
            _Float16* dst = dstT + ((size_t)(b * H_ + hh) * S_ + s) * HD_;
            dst[ln]      = (_Float16)y0;
            dst[16 + ln] = (_Float16)y1;
            dst[32 + ln] = (_Float16)y2;
            dst[48 + ln] = (_Float16)y3;
        }
    }
}

// ---------------------------------------------------------------------------
// 3) Flash attention. 4 waves/block = 4 consecutive q-tiles of the SAME (b,h);
//    each 32-key K/V tile is async-staged into LDS once per block and shared,
//    cutting L2 read traffic 4x. Online softmax; P goes through LDS to convert
//    C-layout -> A-fragment layout.
constexpr int KT_STRIDE = 72;   // halfs per key row (64 + 8 pad, 16B multiple)
constexpr int VT_STRIDE = 40;   // halfs per d row   (32 + 8 pad, 16B multiple)

__global__ void __launch_bounds__(128)
flash_attn_kernel(const _Float16* __restrict__ Q, const _Float16* __restrict__ Kk,
                  const _Float16* __restrict__ Vt, const unsigned char* __restrict__ msk,
                  _Float16* __restrict__ AO) {
    __shared__ _Float16 kt_s[32 * KT_STRIDE];
    __shared__ _Float16 vt_s[64 * VT_STRIDE];
    __shared__ _Float16 pbuf[4][16 * 32];

    const int tid   = threadIdx.x;
    const int lane  = tid & 31;
    const int wave  = tid >> 5;
    const int hi    = lane >> 4;
    const int ln    = lane & 15;
    const int abase = hi ? 8 : 0;
    const int kbase = hi ? 16 : 0;
    const int row0  = hi ? 8 : 0;

    const int tile0 = blockIdx.x * 4;                  // 4 | (S_/16) so same bh
    const int bh    = tile0 / (S_ / 16);
    const int b     = bh / H_;
    const int h     = bh % H_;
    const int qt0   = tile0 & (S_ / 16 - 1);
    const int q0    = (qt0 + wave) * 16;
    const int kmax  = (qt0 + 3) * 16 + 15;             // last wave's causal max

    const _Float16* Qb = Q  + (size_t)bh * S_ * HD_;
    const _Float16* Kb = Kk + (size_t)bh * S_ * HD_;
    const _Float16* Vb = Vt + (size_t)bh * HD_ * S_;
    const unsigned char* mb = msk + (size_t)b * S_;

    // per-thread staging slots: K tile 32x64 halfs, V tile 64x32 halfs
    const int kkey  = tid >> 2;            // 0..31
    const int kpart = (tid & 3) * 16;      // halfs
    const int vd    = tid >> 1;            // 0..63
    const int vpart = (tid & 1) * 16;      // halfs

    v8f o0 = {}, o1 = {}, o2 = {}, o3 = {};
    float mrow[8], lrow[8];
#pragma unroll
    for (int r = 0; r < 8; ++r) { mrow[r] = -3.0e30f; lrow[r] = 0.0f; }

    const _Float16* qrow = Qb + (size_t)(q0 + ln) * HD_;
    _Float16* pb = &pbuf[wave][0];

    for (int k0 = 0; k0 <= kmax; k0 += 32) {
        // ---- async-stage K tile [key][d] and V tile [d][s] into LDS
        {
            const _Float16* ks = Kb + (size_t)(k0 + kkey) * HD_ + kpart;
            const unsigned  kd = lds_off(kt_s + kkey * KT_STRIDE + kpart);
            async_b128(ks,     kd);
            async_b128(ks + 8, kd + 16);
            const _Float16* vs = Vb + (size_t)vd * S_ + k0 + vpart;
            const unsigned  vdo = lds_off(vt_s + vd * VT_STRIDE + vpart);
            async_b128(vs,     vdo);
            async_b128(vs + 8, vdo + 16);
        }
        wait_async0();
        __syncthreads();

        if (k0 <= q0 + 15) {               // wave-uniform causal predicate
            // ---- scores: two 16x16 tiles over 32 keys (scale folded in Q)
            v8f s0 = {}, s1 = {};
#pragma unroll
            for (int dk = 0; dk < HD_; dk += 32) {
                v16h a  = fragA(qrow + dk, abase);
                v16h b0 = fragB(kt_s + (ln)      * KT_STRIDE + dk + kbase);
                v16h b1 = fragB(kt_s + (16 + ln) * KT_STRIDE + dk + kbase);
                s0 = wmma_f16(a, b0, s0);
                s1 = wmma_f16(a, b1, s1);
            }
            // ---- mask + online softmax (lane holds col=ln, rows row0+r)
            const int key0 = k0 + ln, key1 = k0 + 16 + ln;
            const bool pad0 = mb[key0] != 0;
            const bool pad1 = mb[key1] != 0;
#pragma unroll
            for (int r = 0; r < 8; ++r) {
                const int q = q0 + row0 + r;
                float x0 = (pad0 && key0 <= q) ? s0[r] : -3.0e30f;
                float x1 = (pad1 && key1 <= q) ? s1[r] : -3.0e30f;
                float mt = fmaxf(x0, x1);
#pragma unroll
                for (int off = 1; off < 16; off <<= 1)
                    mt = fmaxf(mt, __shfl_xor(mt, off, 32));
                const float mnew  = fmaxf(mrow[r], mt);
                const float alpha = __expf(mrow[r] - mnew);
                const float p0 = __expf(x0 - mnew);
                const float p1 = __expf(x1 - mnew);
                float ps = p0 + p1;
#pragma unroll
                for (int off = 1; off < 16; off <<= 1)
                    ps += __shfl_xor(ps, off, 32);
                lrow[r] = lrow[r] * alpha + ps;
                mrow[r] = mnew;
                o0[r] *= alpha; o1[r] *= alpha; o2[r] *= alpha; o3[r] *= alpha;
                pb[(row0 + r) * 32 + ln]      = (_Float16)p0;
                pb[(row0 + r) * 32 + 16 + ln] = (_Float16)p1;
            }
            // ---- P (16x32) @ V (32x64) from LDS
            v16h pf = fragA(pb + (size_t)ln * 32, abase);
            v16h v0 = fragB(vt_s + (0 * 16 + ln) * VT_STRIDE + kbase);
            v16h v1 = fragB(vt_s + (1 * 16 + ln) * VT_STRIDE + kbase);
            v16h v2 = fragB(vt_s + (2 * 16 + ln) * VT_STRIDE + kbase);
            v16h v3 = fragB(vt_s + (3 * 16 + ln) * VT_STRIDE + kbase);
            o0 = wmma_f16(pf, v0, o0);
            o1 = wmma_f16(pf, v1, o1);
            o2 = wmma_f16(pf, v2, o2);
            o3 = wmma_f16(pf, v3, o3);
        }
        __syncthreads();                   // protect tiles before next stage
    }
    // ---- normalize and write AO[b][s][h*HD + d] (f16 for out-proj GEMM)
#pragma unroll
    for (int r = 0; r < 8; ++r) {
        const float inv = 1.0f / lrow[r];
        _Float16* dst = AO + (size_t)(b * S_ + q0 + row0 + r) * TD_ + h * HD_ + ln;
        dst[0]  = (_Float16)(o0[r] * inv);
        dst[16] = (_Float16)(o1[r] * inv);
        dst[32] = (_Float16)(o2[r] * inv);
        dst[48] = (_Float16)(o3[r] * inv);
    }
}

// ---------------------------------------------------------------------------
// 4) plain GEMM for the output projection: C f32 = A f16 * W f16^T
__global__ void __launch_bounds__(128)
gemm_f16f32_kernel(const _Float16* __restrict__ A, const _Float16* __restrict__ W,
                   float* __restrict__ C, int M, int N, int K) {
    const int lane  = threadIdx.x & 31;
    const int wave  = threadIdx.x >> 5;
    const int hi    = lane >> 4;
    const int ln    = lane & 15;
    const int abase = hi ? 8 : 0;
    const int kbase = hi ? 16 : 0;
    const int m0 = (blockIdx.y * 4 + wave) * 16;
    const int n0 = blockIdx.x * 64;

    v8f acc0 = {}, acc1 = {}, acc2 = {}, acc3 = {};
    gemm_mainloop(A + (size_t)(m0 + ln) * K,
                  W + (size_t)(n0      + ln) * K + kbase,
                  W + (size_t)(n0 + 16 + ln) * K + kbase,
                  W + (size_t)(n0 + 32 + ln) * K + kbase,
                  W + (size_t)(n0 + 48 + ln) * K + kbase,
                  K, abase, acc0, acc1, acc2, acc3);

    const int row0 = hi ? 8 : 0;
#pragma unroll
    for (int r = 0; r < 8; ++r) {
        float* crow = C + (size_t)(m0 + row0 + r) * N + n0 + ln;
        crow[0]  = acc0[r];
        crow[16] = acc1[r];
        crow[32] = acc2[r];
        crow[48] = acc3[r];
    }
}

// ---------------------------------------------------------------------------
extern "C" void kernel_launch(void* const* d_in, const int* in_sizes, int n_in,
                              void* d_out, int out_size, void* d_ws, size_t ws_size,
                              hipStream_t stream) {
    (void)in_sizes; (void)n_in; (void)out_size; (void)ws_size;
    const float* query = (const float*)d_in[0];   // [B,S,DM] f32
    const float* W_in  = (const float*)d_in[1];   // [3*TD,DM] f32
    const float* W_out = (const float*)d_in[2];   // [DM,TD] f32
    const float* sin_q = (const float*)d_in[3];   // [1,1,S,HD] f32
    const float* cos_q = (const float*)d_in[4];   // [1,1,S,HD] f32
    const unsigned char* attn_mask = (const unsigned char*)d_in[5]; // [B,S] bool
    float* out = (float*)d_out;                   // [B,S,DM] f32

    // ---- workspace carve-up (48 MB total)
    char* ws = (char*)d_ws;
    _Float16* qf  = (_Float16*)ws; ws += (size_t)B_ * S_ * DM_ * 2;        // 8 MB
    _Float16* wif = (_Float16*)ws; ws += (size_t)3 * TD_ * DM_ * 2;        // 6 MB
    _Float16* wof = (_Float16*)ws; ws += (size_t)DM_ * TD_ * 2;            // 2 MB
    _Float16* Qr  = (_Float16*)ws; ws += (size_t)B_ * H_ * S_ * HD_ * 2;   // 8 MB
    _Float16* Kr  = (_Float16*)ws; ws += (size_t)B_ * H_ * S_ * HD_ * 2;   // 8 MB
    _Float16* Vt  = (_Float16*)ws; ws += (size_t)B_ * H_ * HD_ * S_ * 2;   // 8 MB
    _Float16* AO  = (_Float16*)ws; ws += (size_t)B_ * S_ * TD_ * 2;        // 8 MB

    // 1) f32 -> f16 conversions
    {
        int n = B_ * S_ * DM_;
        cvt_f16_kernel<<<(n + 255) / 256, 256, 0, stream>>>(query, qf, n);
        n = 3 * TD_ * DM_;
        cvt_f16_kernel<<<(n + 255) / 256, 256, 0, stream>>>(W_in, wif, n);
        n = DM_ * TD_;
        cvt_f16_kernel<<<(n + 255) / 256, 256, 0, stream>>>(W_out, wof, n);
    }
    // 2) fused QKV projection + RoPE + head split (no f32 intermediate)
    gemm_qkv_rope_kernel<<<dim3(3 * TD_ / 64, B_ * S_ / 64), 128, 0, stream>>>(
        qf, wif, sin_q, cos_q, Qr, Kr, Vt);
    // 3) flash attention: 4096 wave-tiles, 4 waves/block sharing K/V via LDS
    flash_attn_kernel<<<(B_ * H_ * (S_ / 16)) / 4, 128, 0, stream>>>(Qr, Kr, Vt, attn_mask, AO);
    // 4) output projection: [B*S, TD] x [TD, DM] -> [B*S, DM] f32
    gemm_f16f32_kernel<<<dim3(DM_ / 64, B_ * S_ / 64), 128, 0, stream>>>(
        AO, wof, out, B_ * S_, DM_, TD_);
}
